// CrossAttentionBlock_32195074851266
// MI455X (gfx1250) — matrix operand order repaired and checked by reading, not compiled, not run on hardware
//
#include <hip/hip_runtime.h>
#include <hip/hip_bf16.h>

typedef __attribute__((ext_vector_type(16))) __bf16 v16bf;
typedef __attribute__((ext_vector_type(8)))  float  v8f;
typedef __attribute__((ext_vector_type(4)))  unsigned int v4u;
typedef __attribute__((ext_vector_type(8)))  int v8i;
typedef __attribute__((ext_vector_type(4)))  int v4i;

static constexpr int BDc   = 64;
static constexpr int CCH   = 5;
static constexpr int HPc   = 112;
static constexpr int WPc   = 112;
static constexpr int EMB   = 192;
static constexpr int HD    = 96;          // head dim
static constexpr int MR    = BDc * 256;   // 16384 rows
static constexpr int NPAN  = 96;          // weight panel columns staged in LDS

__device__ __forceinline__ __bf16 tobf(float f) { return (__bf16)f; }

// ---------------- weight transpose + bf16 convert: Wt[n][k] = W[k][n] ----------------
__global__ void wtrans_kernel(const float* __restrict__ W, __bf16* __restrict__ Wt,
                              int K, int N, int Kpad, int Npad) {
  int i = blockIdx.x * blockDim.x + threadIdx.x;
  if (i >= Kpad * Npad) return;
  int n = i / Kpad, k = i % Kpad;
  float v = (n < N && k < K) ? W[(size_t)k * N + n] : 0.f;
  Wt[(size_t)n * Kpad + k] = tobf(v);
}

// ---------------- patchify: xt[b*256+n][j] (padded to 256 cols) ----------------
__global__ void patchify_kernel(const float* __restrict__ x, float* __restrict__ xt) {
  int i = blockIdx.x * blockDim.x + threadIdx.x;
  if (i >= MR * 256) return;
  int j = i & 255, row = i >> 8;
  int b = row >> 8, n = row & 255;
  float v = 0.f;
  if (j < 245) {
    int c = j / 49, r = j % 49, pi = r / 7, pj = r % 7;
    int p1 = n >> 4, p2 = n & 15;
    v = x[(((size_t)b * CCH + c) * HPc + (p1 * 7 + pi)) * WPc + (p2 * 7 + pj)];
  }
  xt[(size_t)row * 256 + j] = v;
}

// ---------------- gather: yt[b*256+n][c*64+v] ----------------
__global__ void gather_kernel(const float* __restrict__ y, const int* __restrict__ idx0,
                              const int* __restrict__ idx1, float* __restrict__ yt) {
  int i = blockIdx.x * blockDim.x + threadIdx.x;
  if (i >= MR * 320) return;
  int j = i % 320; int row = i / 320;
  int b = row >> 8, n = row & 255;
  int c = j >> 6, v = j & 63;
  int a  = idx0[n * 64 + v];
  int bb = idx1[n * 64 + v];
  yt[(size_t)row * 320 + j] = y[(((size_t)b * CCH + c) * 192 + a) * 256 + bb];
}

// ---------------- TDM-staged WMMA GEMM ----------------
// Block = 8 waves (256 thr). Block computes 128 rows x 96 cols; lda == KPAD for all calls.
// Weight panel (96 cols x KPAD) is DMA'd to LDS by the Tensor Data Mover once per block.
// grid = (ceil(Nc/96), MR/128)
template<int KPAD, bool HAS_BIAS, bool HAS_POS, int ACT>
__global__ __launch_bounds__(256, 1)
void gemm_tdm_kernel(const float* __restrict__ A, const __bf16* __restrict__ Wt, int Npad,
                     const float* __restrict__ bias, const float* __restrict__ pos,
                     float* __restrict__ out, int ldo, int Nc) {
  __shared__ __bf16 wlds[NPAN * KPAD];    // 36/48/60 KB
  int lane  = threadIdx.x & 31;
  int wave  = threadIdx.x >> 5;
  int nbase = blockIdx.x * NPAN;
  int mbase = blockIdx.y * 128 + wave * 16;

  // --- stage weight panel into LDS via Tensor Data Mover (wave 0 only) ---
#if __has_builtin(__builtin_amdgcn_tensor_load_to_lds)
  if (wave == 0) {
    unsigned long long ga = (unsigned long long)(uintptr_t)(Wt + (size_t)nbase * KPAD);
    unsigned int lds_off  = (unsigned int)(uintptr_t)(&wlds[0]);
    v4u g0; v8i g1; v4i gz = {0, 0, 0, 0};
    g0[0] = 1u;                                     // count=1, user descriptor
    g0[1] = lds_off;                                // lds_addr (bytes)
    g0[2] = (unsigned int)(ga & 0xffffffffu);       // global_addr[31:0]
    g0[3] = (unsigned int)((ga >> 32) & 0x01ffffffu) | 0x80000000u; // addr[56:32] | type=2
    g1[0] = (int)(1u << 16);                        // data_size=1 (2 bytes), no flags
    g1[1] = (int)((unsigned)(KPAD & 0xffff) << 16); // tensor_dim0[15:0] @ bits63:48
    g1[2] = (int)((unsigned)(Npad & 0xffff) << 16); // tensor_dim0 hi=0; tensor_dim1[15:0]
    g1[3] = (int)((unsigned)(KPAD & 0xffff) << 16); // tensor_dim1 hi=0; tile_dim0=KPAD
    g1[4] = (int)(unsigned)NPAN;                    // tile_dim1=96; tile_dim2=0
    g1[5] = KPAD;                                   // tensor_dim0_stride lo
    g1[6] = 0;                                      // stride hi / dim1_stride lo
    g1[7] = 0;
#if defined(__clang_major__) && __clang_major__ >= 23
    v8i gz8 = {0, 0, 0, 0, 0, 0, 0, 0};
    __builtin_amdgcn_tensor_load_to_lds(g0, g1, gz, gz, gz8, 0);
#else
    __builtin_amdgcn_tensor_load_to_lds(g0, g1, gz, gz, 0);
#endif
    __builtin_amdgcn_s_wait_tensorcnt(0);
  }
  __syncthreads();
#else
  for (int i = threadIdx.x; i < NPAN * KPAD; i += 256) {
    int n = i / KPAD, k = i % KPAD;
    wlds[i] = ((nbase + n) < Npad) ? Wt[(size_t)(nbase + n) * KPAD + k] : (__bf16)0.f;
  }
  __syncthreads();
#endif

  int aoff = (lane >> 4) << 3;   // 0 / 8   (A-fragment K offset)
  int boff = (lane >> 4) << 3;   // 0 / 8   (B-fragment K offset)
  const float*  arow  = A + (size_t)(mbase + (lane & 15)) * KPAD + aoff;
  const __bf16* bbase = wlds + (lane & 15) * KPAD + boff;

  v8f acc[6];
#pragma unroll
  for (int ct = 0; ct < 6; ++ct) acc[ct] = (v8f){};

#pragma unroll
  for (int k = 0; k < KPAD; k += 32) {
    v16bf af;
#pragma unroll
    for (int hh = 0; hh < 8; ++hh) {
      af[hh]     = tobf(arow[k + hh]);
      af[hh + 8] = tobf(arow[k + 16 + hh]);
    }
#pragma unroll
    for (int ct = 0; ct < 6; ++ct) {
      const __bf16* bp = bbase + ct * 16 * KPAD + k;   // compile-time tile offsets
      v16bf bf;
#pragma unroll
      for (int hh = 0; hh < 8; ++hh) { bf[hh] = bp[hh]; bf[hh + 8] = bp[16 + hh]; }
      acc[ct] = __builtin_amdgcn_wmma_f32_16x16x32_bf16(false, af, false, bf, (short)0, acc[ct], false, false);
    }
  }

  // C layout: vgpr r, lane L -> row = r + (L>=16)*8, col = L&15
  int mo = mbase + ((lane >> 4) << 3);
#pragma unroll
  for (int ct = 0; ct < 6; ++ct) {
    int n = nbase + ct * 16 + (lane & 15);
    if (n < Nc) {
#pragma unroll
      for (int r = 0; r < 8; ++r) {
        int m = mo + r;
        float v = acc[ct][r];
        if (HAS_BIAS) v += bias[n];
        if (HAS_POS)  v += pos[(size_t)(m & 255) * EMB + n];
        if (ACT == 1) v = 0.5f * v * (1.f + erff(v * 0.70710678118f));
        out[(size_t)m * ldo + n] = v;
      }
    }
  }
}

// ---------------- Vt[b][h][d][n] = bf16(V[(b*256+n)][h*96+d]) ----------------
__global__ void vtrans_kernel(const float* __restrict__ Vb, __bf16* __restrict__ Vt) {
  int i = blockIdx.x * blockDim.x + threadIdx.x;
  if (i >= BDc * 2 * HD * 256) return;
  int n = i & 255, rest = i >> 8;
  int d = rest % HD, bh = rest / HD;
  int h = bh & 1, b = bh >> 1;
  Vt[i] = tobf(Vb[(size_t)(b * 256 + n) * EMB + h * HD + d]);
}

// ---------------- cross-attention: out[b,n,h*96+d] = softmax(Q K^T / sqrt(96)) V ----------------
__global__ __launch_bounds__(128, 1)
void attn_kernel(const float* __restrict__ Q, const float* __restrict__ Kb,
                 const __bf16* __restrict__ Vt, float* __restrict__ out) {
  __shared__ __bf16 p_lds[4][16][256];
  int lane = threadIdx.x & 31;
  int wave = threadIdx.x >> 5;
  int rb = blockIdx.x & 3;
  int h  = (blockIdx.x >> 2) & 1;
  int b  = blockIdx.x >> 3;
  int mbase = rb * 64 + wave * 16;
  int aoff  = (lane >> 4) << 3;
  int boff  = (lane >> 4) << 3;
  int half8 = (lane >> 4) << 3;
  const float scale = 0.10206207261596575f;   // 96^-0.5

  // per-lane base pointers; all tile offsets below are compile-time immediates
  const float*  qbase = Q  + (size_t)(b * 256 + mbase + (lane & 15)) * EMB + h * HD + aoff;
  const float*  kbase = Kb + (size_t)(b * 256 + (lane & 15)) * EMB + h * HD + boff;
  const __bf16* vbase = Vt + ((size_t)(b * 2 + h) * HD + (lane & 15)) * 256 + boff;
  float*        obase = out + (size_t)(b * 256 + mbase + half8) * EMB + h * HD + (lane & 15);

  v8f s[16];
#pragma unroll
  for (int t = 0; t < 16; ++t) s[t] = (v8f){};

#pragma unroll
  for (int kt = 0; kt < 3; ++kt) {            // head_dim 96 = 3 * 32
    v16bf af;                                  // fold 1/sqrt(96) into Q fragment
#pragma unroll
    for (int hh = 0; hh < 8; ++hh) {
      af[hh]     = tobf(qbase[kt * 32 + hh] * scale);
      af[hh + 8] = tobf(qbase[kt * 32 + 16 + hh] * scale);
    }
#pragma unroll
    for (int nt = 0; nt < 16; ++nt) {         // 256 keys = 16 col tiles
      v16bf bf;
#pragma unroll
      for (int hh = 0; hh < 8; ++hh) { bf[hh] = tobf(kbase[nt * 16 * EMB + kt * 32 + hh]); bf[hh + 8] = tobf(kbase[nt * 16 * EMB + kt * 32 + 16 + hh]); }
      s[nt] = __builtin_amdgcn_wmma_f32_16x16x32_bf16(false, af, false, bf, (short)0, s[nt], false, false);
    }
  }

#pragma unroll
  for (int r = 0; r < 8; ++r) {               // softmax: row r+half8, cols across 16 lanes of half
    float mx = -1e30f;
#pragma unroll
    for (int t = 0; t < 16; ++t) mx = fmaxf(mx, s[t][r]);
#pragma unroll
    for (int d = 1; d < 16; d <<= 1) mx = fmaxf(mx, __shfl_xor(mx, d, 32));
    float sum = 0.f;
#pragma unroll
    for (int t = 0; t < 16; ++t) { float e = __expf(s[t][r] - mx); s[t][r] = e; sum += e; }
#pragma unroll
    for (int d = 1; d < 16; d <<= 1) sum += __shfl_xor(sum, d, 32);
    float inv = 1.f / sum;
#pragma unroll
    for (int t = 0; t < 16; ++t) s[t][r] *= inv;
  }

#pragma unroll
  for (int r = 0; r < 8; ++r) {
    int row = half8 + r;
#pragma unroll
    for (int t = 0; t < 16; ++t)
      p_lds[wave][row][t * 16 + (lane & 15)] = tobf(s[t][r]);
  }
  __syncthreads();

  v8f o[6];
#pragma unroll
  for (int ct = 0; ct < 6; ++ct) o[ct] = (v8f){};
  const __bf16* pbase = &p_lds[wave][lane & 15][aoff];
#pragma unroll
  for (int kt = 0; kt < 8; ++kt) {            // 256 keys = 8 * 32 K-steps
    v16bf af;
#pragma unroll
    for (int hh = 0; hh < 8; ++hh) {
      af[hh]     = pbase[kt * 32 + hh];
      af[hh + 8] = pbase[kt * 32 + 16 + hh];
    }
#pragma unroll
    for (int ct = 0; ct < 6; ++ct) {          // head_dim 96 = 6 col tiles
      v16bf bf;
#pragma unroll
      for (int hh = 0; hh < 8; ++hh) { bf[hh] = vbase[ct * 16 * 256 + kt * 32 + hh]; bf[hh + 8] = vbase[ct * 16 * 256 + kt * 32 + 16 + hh]; }
      o[ct] = __builtin_amdgcn_wmma_f32_16x16x32_bf16(false, af, false, bf, (short)0, o[ct], false, false);
    }
  }
#pragma unroll
  for (int ct = 0; ct < 6; ++ct)
#pragma unroll
    for (int r = 0; r < 8; ++r)
      obase[r * EMB + ct * 16] = o[ct][r];
}

// ---------------- out = LayerNorm(a + r) * g + be ; 192 threads per row ----------------
__global__ void add_ln_kernel(const float* __restrict__ a, const float* __restrict__ r,
                              const float* __restrict__ g, const float* __restrict__ be,
                              float* __restrict__ out) {
  __shared__ float red[192];
  __shared__ float sm[2];
  int row = blockIdx.x, t = threadIdx.x;
  float v = a[(size_t)row * EMB + t] + r[(size_t)row * EMB + t];
  red[t] = v; __syncthreads();
  for (int s2 = 128; s2 > 0; s2 >>= 1) { if (t < s2 && t + s2 < 192) red[t] += red[t + s2]; __syncthreads(); }
  if (t == 0) sm[0] = red[0] * (1.f / 192.f);
  __syncthreads();
  float d = v - sm[0];
  red[t] = d * d; __syncthreads();
  for (int s2 = 128; s2 > 0; s2 >>= 1) { if (t < s2 && t + s2 < 192) red[t] += red[t + s2]; __syncthreads(); }
  if (t == 0) sm[1] = red[0] * (1.f / 192.f);
  __syncthreads();
  out[(size_t)row * EMB + t] = d * rsqrtf(sm[1] + 1e-5f) * g[t] + be[t];
}

// ---------------- epilogues ----------------
__global__ void unpatch_kernel(const float* __restrict__ x, const float* __restrict__ xun,
                               float* __restrict__ outx) {
  int i = blockIdx.x * blockDim.x + threadIdx.x;
  if (i >= BDc * HPc * WPc) return;
  int wp = i % WPc, rest = i / WPc, hp = rest % HPc, b = rest / HPc;
  int p1 = hp / 7, pi = hp % 7, p2 = wp / 7, pj = wp % 7;
  int n = p1 * 16 + p2, j = pi * 7 + pj;
  outx[i] = x[(((size_t)b * CCH + 4) * HPc + hp) * WPc + wp] + xun[(size_t)(b * 256 + n) * 49 + j];
}

__global__ void ycopy_kernel(const float* __restrict__ y, float* __restrict__ outy) {
  int i = blockIdx.x * blockDim.x + threadIdx.x;
  if (i >= BDc * 192 * 256) return;
  int r = i % (192 * 256), b = i / (192 * 256);
  outy[i] = y[((size_t)b * CCH + 4) * 49152 + r];
}

__global__ void yscatter_kernel(const float* __restrict__ yc, const float* __restrict__ w,
                                const int* __restrict__ idx0, const int* __restrict__ idx1,
                                float* __restrict__ outy) {
  int i = blockIdx.x * blockDim.x + threadIdx.x;
  if (i >= BDc * 256 * 64) return;
  int v = i & 63, rest = i >> 6, n = rest & 255, b = rest >> 8;
  int a  = idx0[n * 64 + v];
  int bb = idx1[n * 64 + v];
  outy[(size_t)b * 49152 + a * 256 + bb] += yc[((size_t)b * 256 + n) * 64 + v] * w[n * 64 + v];
}

// =======================================================================================
extern "C" void kernel_launch(void* const* d_in, const int* in_sizes, int n_in,
                              void* d_out, int out_size, void* d_ws, size_t ws_size,
                              hipStream_t stream) {
  const float* x    = (const float*)d_in[0];
  const float* y    = (const float*)d_in[1];
  const int*   idx0 = (const int*)d_in[2];
  const int*   idx1 = (const int*)d_in[3];
  const float* wsc  = (const float*)d_in[4];
  const float* W_emb_p = (const float*)d_in[5];
  const float* b_emb_p = (const float*)d_in[6];
  const float* W_emb_d = (const float*)d_in[7];
  const float* b_emb_d = (const float*)d_in[8];
  const float* pos_p = (const float*)d_in[9];
  const float* pos_d = (const float*)d_in[10];
  const float* Wq = (const float*)d_in[11];
  const float* Wk = (const float*)d_in[12];
  const float* Wv = (const float*)d_in[13];
  const float* Wmp[3] = {(const float*)d_in[14], (const float*)d_in[16], (const float*)d_in[18]};
  const float* bmp[3] = {(const float*)d_in[15], (const float*)d_in[17], (const float*)d_in[19]};
  const float* Wmd[3] = {(const float*)d_in[20], (const float*)d_in[22], (const float*)d_in[24]};
  const float* bmd[3] = {(const float*)d_in[21], (const float*)d_in[23], (const float*)d_in[25]};
  const float* W_un_p = (const float*)d_in[26];
  const float* b_un_p = (const float*)d_in[27];
  const float* W_un_d = (const float*)d_in[28];
  const float* b_un_d = (const float*)d_in[29];
  const float* g_att_p = (const float*)d_in[30]; const float* be_att_p = (const float*)d_in[31];
  const float* g_att_d = (const float*)d_in[32]; const float* be_att_d = (const float*)d_in[33];
  const float* g_mlp_p = (const float*)d_in[34]; const float* be_mlp_p = (const float*)d_in[35];
  const float* g_mlp_d = (const float*)d_in[36]; const float* be_mlp_d = (const float*)d_in[37];

  float* out_x = (float*)d_out;
  float* out_y = (float*)d_out + (size_t)BDc * HPc * WPc;

  // ---- workspace carve ----
  char* p = (char*)d_ws;
  auto alloc = [&](size_t bytes) { void* r = (void*)p; p += (bytes + 255) & ~(size_t)255; return r; };
  const size_t SZE = (size_t)MR * EMB * sizeof(float);
  float* xt   = (float*)alloc((size_t)MR * 256 * 4);
  float* yt   = (float*)alloc((size_t)MR * 320 * 4);
  float* xemb = (float*)alloc(SZE);
  float* yemb = (float*)alloc(SZE);
  float* Qp = (float*)alloc(SZE); float* Kp = (float*)alloc(SZE); float* Vp = (float*)alloc(SZE);
  float* Qd = (float*)alloc(SZE); float* Kd = (float*)alloc(SZE); float* Vd = (float*)alloc(SZE);
  __bf16* Vt_p = (__bf16*)alloc((size_t)BDc * 2 * HD * 256 * 2);
  __bf16* Vt_d = (__bf16*)alloc((size_t)BDc * 2 * HD * 256 * 2);
  __bf16* wt_emb_p = (__bf16*)alloc((size_t)192 * 256 * 2);
  __bf16* wt_emb_d = (__bf16*)alloc((size_t)192 * 320 * 2);
  __bf16* wt_q = (__bf16*)alloc((size_t)192 * 192 * 2);
  __bf16* wt_k = (__bf16*)alloc((size_t)192 * 192 * 2);
  __bf16* wt_v = (__bf16*)alloc((size_t)192 * 192 * 2);
  __bf16* wt_mp[3]; __bf16* wt_md[3];
  for (int i = 0; i < 3; ++i) wt_mp[i] = (__bf16*)alloc((size_t)192 * 192 * 2);
  for (int i = 0; i < 3; ++i) wt_md[i] = (__bf16*)alloc((size_t)192 * 192 * 2);
  __bf16* wt_un_p = (__bf16*)alloc((size_t)64 * 192 * 2);
  __bf16* wt_un_d = (__bf16*)alloc((size_t)64 * 192 * 2);
  float* attp = xt;   float* attd = yt;     // reuse dead buffers
  float* xun  = Kp;   float* yc   = Kd;     // valid after MLP temps are done

  auto WT = [&](const float* W, __bf16* Wt, int K, int N, int Kpad, int Npad) {
    int tot = Kpad * Npad;
    wtrans_kernel<<<(tot + 255) / 256, 256, 0, stream>>>(W, Wt, K, N, Kpad, Npad);
  };
  WT(W_emb_p, wt_emb_p, 245, 192, 256, 192);
  WT(W_emb_d, wt_emb_d, 320, 192, 320, 192);
  WT(Wq, wt_q, 192, 192, 192, 192);
  WT(Wk, wt_k, 192, 192, 192, 192);
  WT(Wv, wt_v, 192, 192, 192, 192);
  for (int i = 0; i < 3; ++i) WT(Wmp[i], wt_mp[i], 192, 192, 192, 192);
  for (int i = 0; i < 3; ++i) WT(Wmd[i], wt_md[i], 192, 192, 192, 192);
  WT(W_un_p, wt_un_p, 192, 49, 192, 64);
  WT(W_un_d, wt_un_d, 192, 64, 192, 64);

  patchify_kernel<<<(MR * 256 + 255) / 256, 256, 0, stream>>>(x, xt);
  gather_kernel<<<(MR * 320 + 255) / 256, 256, 0, stream>>>(y, idx0, idx1, yt);

  dim3 g2(2, MR / 128), g1(1, MR / 128);
  // embeddings: bias + pos
  gemm_tdm_kernel<256, true, true, 0><<<g2, 256, 0, stream>>>(xt, wt_emb_p, 192, b_emb_p, pos_p, xemb, EMB, EMB);
  gemm_tdm_kernel<320, true, true, 0><<<g2, 256, 0, stream>>>(yt, wt_emb_d, 192, b_emb_d, pos_d, yemb, EMB, EMB);
  // QKV: plain
  gemm_tdm_kernel<192, false, false, 0><<<g2, 256, 0, stream>>>(xemb, wt_q, 192, nullptr, nullptr, Qp, EMB, EMB);
  gemm_tdm_kernel<192, false, false, 0><<<g2, 256, 0, stream>>>(xemb, wt_k, 192, nullptr, nullptr, Kp, EMB, EMB);
  gemm_tdm_kernel<192, false, false, 0><<<g2, 256, 0, stream>>>(xemb, wt_v, 192, nullptr, nullptr, Vp, EMB, EMB);
  gemm_tdm_kernel<192, false, false, 0><<<g2, 256, 0, stream>>>(yemb, wt_q, 192, nullptr, nullptr, Qd, EMB, EMB);
  gemm_tdm_kernel<192, false, false, 0><<<g2, 256, 0, stream>>>(yemb, wt_k, 192, nullptr, nullptr, Kd, EMB, EMB);
  gemm_tdm_kernel<192, false, false, 0><<<g2, 256, 0, stream>>>(yemb, wt_v, 192, nullptr, nullptr, Vd, EMB, EMB);

  int vtot = BDc * 2 * HD * 256;
  vtrans_kernel<<<(vtot + 255) / 256, 256, 0, stream>>>(Vp, Vt_p);
  vtrans_kernel<<<(vtot + 255) / 256, 256, 0, stream>>>(Vd, Vt_d);

  attn_kernel<<<BDc * 2 * 4, 128, 0, stream>>>(Qd, Kp, Vt_p, attp);  // att_p
  attn_kernel<<<BDc * 2 * 4, 128, 0, stream>>>(Qp, Kd, Vt_d, attd);  // att_d

  add_ln_kernel<<<MR, 192, 0, stream>>>(xemb, attp, g_att_p, be_att_p, xemb);
  add_ln_kernel<<<MR, 192, 0, stream>>>(yemb, attd, g_att_d, be_att_d, yemb);

  // MLP p (temps Qp/Kp), bias + exact GELU
  gemm_tdm_kernel<192, true, false, 1><<<g2, 256, 0, stream>>>(xemb, wt_mp[0], 192, bmp[0], nullptr, Qp, EMB, EMB);
  gemm_tdm_kernel<192, true, false, 1><<<g2, 256, 0, stream>>>(Qp,   wt_mp[1], 192, bmp[1], nullptr, Kp, EMB, EMB);
  gemm_tdm_kernel<192, true, false, 1><<<g2, 256, 0, stream>>>(Kp,   wt_mp[2], 192, bmp[2], nullptr, Qp, EMB, EMB);
  add_ln_kernel<<<MR, 192, 0, stream>>>(xemb, Qp, g_mlp_p, be_mlp_p, xemb);
  // MLP d (temps Qd/Kd)
  gemm_tdm_kernel<192, true, false, 1><<<g2, 256, 0, stream>>>(yemb, wt_md[0], 192, bmd[0], nullptr, Qd, EMB, EMB);
  gemm_tdm_kernel<192, true, false, 1><<<g2, 256, 0, stream>>>(Qd,   wt_md[1], 192, bmd[1], nullptr, Kd, EMB, EMB);
  gemm_tdm_kernel<192, true, false, 1><<<g2, 256, 0, stream>>>(Kd,   wt_md[2], 192, bmd[2], nullptr, Qd, EMB, EMB);
  add_ln_kernel<<<MR, 192, 0, stream>>>(yemb, Qd, g_mlp_d, be_mlp_d, yemb);

  // un-embeddings
  gemm_tdm_kernel<192, true, false, 0><<<g1, 256, 0, stream>>>(xemb, wt_un_p, 64, b_un_p, nullptr, xun, 49, 49);
  gemm_tdm_kernel<192, true, false, 0><<<g1, 256, 0, stream>>>(yemb, wt_un_d, 64, b_un_d, nullptr, yc, 64, 64);

  unpatch_kernel<<<(BDc * HPc * WPc + 255) / 256, 256, 0, stream>>>(x, xun, out_x);
  ycopy_kernel<<<(BDc * 192 * 256 + 255) / 256, 256, 0, stream>>>(y, out_y);
  yscatter_kernel<<<(BDc * 256 * 64 + 255) / 256, 256, 0, stream>>>(yc, wsc, idx0, idx1, out_y);
}
